// MultiScaleRetention_10548439679589
// MI455X (gfx1250) — matrix operand order repaired and checked by reading, not compiled, not run on hardware
//
#include <hip/hip_runtime.h>
#include <math.h>

// ---------------------------------------------------------------------------
// MultiScaleRetention forward for MI455X (gfx1250, wave32, WMMA bf16)
// ---------------------------------------------------------------------------
#define B_ 2
#define T_ 2048
#define HID_ 1024
#define NH_ 8
#define DQK_ 128
#define DV_ 256
#define VDIM_ 2048
#define QKDIM_ 1024

typedef __attribute__((ext_vector_type(16))) __bf16 bf16x16;
typedef __attribute__((ext_vector_type(8)))  float  f32x8;

union Frag { uint4 q[2]; bf16x16 v; };
union U16x8 { uint4 u4; unsigned short s[8]; };

__device__ __forceinline__ unsigned short f2bf(float f) {
    unsigned int u = __float_as_uint(f);
    unsigned int r = (u + 0x7fffu + ((u >> 16) & 1u)) >> 16;  // RNE
    return (unsigned short)r;
}
__device__ __forceinline__ float bf2f(unsigned short s) {
    return __uint_as_float(((unsigned int)s) << 16);
}
__device__ __forceinline__ f32x8 zero8() {
    f32x8 z;
#pragma unroll
    for (int i = 0; i < 8; ++i) z[i] = 0.0f;
    return z;
}
__device__ __forceinline__ f32x8 wmma_bf16(const bf16x16& a, const bf16x16& b, const f32x8& c) {
    return __builtin_amdgcn_wmma_f32_16x16x32_bf16(false, a, false, b, (short)0, c, false, false);
}

// ---------------------------------------------------------------------------
// K1: fp32 -> bf16 conversion
// ---------------------------------------------------------------------------
__global__ __launch_bounds__(256) void cvt_kernel(const float* __restrict__ in,
                                                  unsigned short* __restrict__ out, int n) {
    int i = blockIdx.x * 256 + threadIdx.x;
    if (i < n) out[i] = f2bf(in[i]);
}

// ---------------------------------------------------------------------------
// K2: generic bf16 WMMA GEMM, C[M,N] = A[M,K] @ B[K,N]  (row-major)
// mode 0: Cf = acc
// mode 1: Cb = bf16( silu(acc) * aux[row,col] )        (gate fused)
// mode 2: Cf = acc + aux[col]                          (bias fused)
// block tile 128x128, 8 waves (2x4), wave tile 64x32, K chunk 32,
// double-buffered LDS, software pipelined; staging values are loop-local
// named uint4s (no aggregates / address-taken objects -> no scratch spills).
// ---------------------------------------------------------------------------
#define GLS 40  // padded LDS K-stride (keeps 16B alignment: 40*2=80 bytes)

__global__ __launch_bounds__(256, 1)
void gemm_bf16_kernel(const unsigned short* __restrict__ A, const unsigned short* __restrict__ Bm,
                      int M, int N, int K, int mode, const float* __restrict__ aux,
                      float* __restrict__ Cf, unsigned short* __restrict__ Cb) {
    __shared__ unsigned short As[2][128 * GLS];
    __shared__ unsigned short Bs[2][128 * GLS];
    const int tid  = threadIdx.x;
    const int lane = tid & 31;
    const int w    = tid >> 5;
    const int wm   = w >> 2, wn = w & 3;
    const int m0   = wm * 64, n0 = wn * 32;
    const int bm   = blockIdx.y * 128;
    const int bn   = blockIdx.x * 128;
    const int ln   = lane & 15;
    const int hiA  = (lane >> 4) * 8;   // A K-half offset (ISA 16-bit A layout)
    const int hiB  = (lane >> 4) * 8;   // B K-half offset (ISA 16-bit B layout)
    const int rA   = tid >> 2, cA = (tid & 3) * 8;   // A staging coords
    const int rB   = tid >> 4, cB = (tid & 15) * 8;  // B staging coords

    f32x8 acc[4][2];
#pragma unroll
    for (int mt = 0; mt < 4; ++mt)
#pragma unroll
        for (int nt = 0; nt < 2; ++nt) acc[mt][nt] = zero8();

    {   // prologue: stage chunk 0 into buffer 0
        uint4 a0 = *(const uint4*)(A + (size_t)(bm + rA) * K + cA);
        uint4 a1 = *(const uint4*)(A + (size_t)(bm + rA + 64) * K + cA);
        U16x8 b0, b1;
        b0.u4 = *(const uint4*)(Bm + (size_t)rB * N + bn + cB);
        b1.u4 = *(const uint4*)(Bm + (size_t)(rB + 16) * N + bn + cB);
        *(uint4*)&As[0][rA * GLS + cA]        = a0;
        *(uint4*)&As[0][(rA + 64) * GLS + cA] = a1;
#pragma unroll
        for (int q = 0; q < 8; ++q) Bs[0][(cB + q) * GLS + rB]      = b0.s[q];  // -> [n][k]
#pragma unroll
        for (int q = 0; q < 8; ++q) Bs[0][(cB + q) * GLS + rB + 16] = b1.s[q];
    }
    __syncthreads();

    int buf = 0;
    for (int k0 = 0; k0 < K; k0 += 32) {
        const bool more = (k0 + 32 < K);
        uint4 na0, na1, nb0, nb1;
        if (more) {  // issue next chunk's global loads early; no waits until stores
            na0 = *(const uint4*)(A + (size_t)(bm + rA) * K + k0 + 32 + cA);
            na1 = *(const uint4*)(A + (size_t)(bm + rA + 64) * K + k0 + 32 + cA);
            nb0 = *(const uint4*)(Bm + (size_t)(k0 + 32 + rB) * N + bn + cB);
            nb1 = *(const uint4*)(Bm + (size_t)(k0 + 32 + rB + 16) * N + bn + cB);
            __builtin_prefetch(A + (size_t)(bm + (tid >> 1)) * K + k0 + 64, 0, 0);
            __builtin_prefetch(Bm + (size_t)(k0 + 64 + lane) * N + bn, 0, 0);
        }
        // compute current chunk from LDS[buf]
        Frag bf[2];
#pragma unroll
        for (int nt = 0; nt < 2; ++nt) {
            int n = n0 + nt * 16 + ln;
            bf[nt].q[0] = *(uint4*)&Bs[buf][n * GLS + hiB];
            bf[nt].q[1] = *(uint4*)&Bs[buf][n * GLS + hiB + 16];
        }
#pragma unroll
        for (int mt = 0; mt < 4; ++mt) {
            int m = m0 + mt * 16 + ln;
            Frag af;
            af.q[0] = *(uint4*)&As[buf][m * GLS + hiA];
            af.q[1] = *(uint4*)&As[buf][m * GLS + 16 + hiA];
#pragma unroll
            for (int nt = 0; nt < 2; ++nt) acc[mt][nt] = wmma_bf16(af.v, bf[nt].v, acc[mt][nt]);
        }
        if (more) {  // drain staged regs into the other LDS buffer
            *(uint4*)&As[buf ^ 1][rA * GLS + cA]        = na0;
            *(uint4*)&As[buf ^ 1][(rA + 64) * GLS + cA] = na1;
            U16x8 u0, u1;
            u0.u4 = nb0; u1.u4 = nb1;
#pragma unroll
            for (int q = 0; q < 8; ++q) Bs[buf ^ 1][(cB + q) * GLS + rB]      = u0.s[q];
#pragma unroll
            for (int q = 0; q < 8; ++q) Bs[buf ^ 1][(cB + q) * GLS + rB + 16] = u1.s[q];
        }
        __syncthreads();
        buf ^= 1;
    }

    const int mb = (lane >> 4) * 8;  // C/D layout: lanes 16-31 hold M+8
#pragma unroll
    for (int mt = 0; mt < 4; ++mt)
#pragma unroll
        for (int nt = 0; nt < 2; ++nt)
#pragma unroll
            for (int v = 0; v < 8; ++v) {
                int row = bm + m0 + mt * 16 + mb + v;
                int col = bn + n0 + nt * 16 + ln;
                float val = acc[mt][nt][v];
                if (mode == 0) {
                    Cf[(size_t)row * N + col] = val;
                } else if (mode == 1) {
                    float g = val / (1.0f + __expf(-val));      // silu
                    Cb[(size_t)row * N + col] = f2bf(g * aux[(size_t)row * N + col]);
                } else {
                    Cf[(size_t)row * N + col] = val + aux[col];
                }
            }
}

// ---------------------------------------------------------------------------
// K3: xPos rotate + split/pack qkv(fp32) -> per-head bf16 q/k/v
// ---------------------------------------------------------------------------
__global__ __launch_bounds__(256)
void rotate_kernel(const float* __restrict__ qkv, unsigned short* __restrict__ qro,
                   unsigned short* __restrict__ kro, unsigned short* __restrict__ vbuf) {
    size_t idx = (size_t)blockIdx.x * 256 + threadIdx.x;  // B*T*2048 threads
    int c2 = (int)(idx & 2047);
    size_t bt = idx >> 11;
    int t = (int)(bt & 2047);
    int b = (int)(bt >> 11);
    const float* row = qkv + bt * 4096;
    if (c2 < 1024) {  // rotary pair of q (c2<512) or k
        int isK = (c2 >= 512);
        int i = c2 & 511;
        float x1 = row[isK * 1024 + 2 * i];
        float x2 = row[isK * 1024 + 2 * i + 1];
        float bscale = ((float)(2 * i) + 0.4f * 1024.0f) / (1.4f * 1024.0f);
        float lsc = __logf(bscale) * ((float)t / 512.0f);
        float scale = __expf(isK ? -lsc : lsc);                    // downscale for k
        float invfreq = __expf(-((float)i / 512.0f) * 9.210340371976184f);  // 10000^-i/dh
        float su = (float)t * invfreq;
        float s, c;
        __sincosf(su, &s, &c);
        s *= scale; c *= scale;
        float y1 = x1 * c - x2 * s;
        float y2 = x2 * c + x1 * s;
        if (!isK) { y1 *= 0.08838834764831845f; y2 *= 0.08838834764831845f; }  // dqk^-0.5 into q
        int f = 2 * i, hh = f >> 7, d = f & 127;
        unsigned short* dst = (isK ? kro : qro) + (((size_t)(b * NH_ + hh) * T_) + t) * DQK_ + d;
        dst[0] = f2bf(y1); dst[1] = f2bf(y2);
    } else {  // v copy
        int j = c2 - 1024, f = 2 * j;
        int hh = f >> 8, d = f & 255;
        unsigned short* dst = vbuf + (((size_t)(b * NH_ + hh) * T_) + t) * DV_ + d;
        dst[0] = f2bf(row[2048 + f]);
        dst[1] = f2bf(row[2048 + f + 1]);
    }
}

__device__ __forceinline__ float head_log2gamma(int h) {
    float lg0 = logf(1.0f / 32.0f), lg1 = logf(1.0f / 512.0f);
    float gamma = 1.0f - __expf(lg0 + (lg1 - lg0) * (float)h / 7.0f);
    return log2f(gamma);
}

// ---------------------------------------------------------------------------
// K4: causal decayed retention: ret[b,t,h*256+n] = sum_j<=i g^(i-j) (q.k) v
// block: one (b,h) x 64-row q tile; j tiles of 32; flash-style, WMMA both matmuls
// ---------------------------------------------------------------------------
__global__ __launch_bounds__(256, 1)
void retention_kernel(const unsigned short* __restrict__ qro, const unsigned short* __restrict__ kro,
                      const unsigned short* __restrict__ vb, float* __restrict__ ret) {
    __shared__ unsigned short Qs[64 * 136];   // q tile [64][128]
    __shared__ unsigned short Ks[32 * 136];   // k tile [32][128] (== Bt layout for S)
    __shared__ unsigned short Vt[256 * GLS];  // v tile transposed [256][32]
    __shared__ unsigned short Ps[64 * GLS];   // decayed probs bf16 [64][32]
    const int tid = threadIdx.x, lane = tid & 31, w = tid >> 5;
    const int ln = lane & 15;
    const int hiA = (lane >> 4) * 8, hiB = (lane >> 4) * 8, mb = (lane >> 4) * 8;
    const int i0 = blockIdx.x * 64;
    const int bh = blockIdx.y, b = bh >> 3, h = bh & 7;
    const float l2g = head_log2gamma(h);
    const size_t hbase = (size_t)(b * NH_ + h) * T_;
    const unsigned short* qh = qro + hbase * DQK_;
    const unsigned short* kh = kro + hbase * DQK_;
    const unsigned short* vh = vb + hbase * DV_;

    {   // stage Q once: 64x128, batched loads then stores
        const int r = tid >> 4, c8 = (tid & 15) * 8;
        uint4 q0 = *(const uint4*)(qh + (size_t)(i0 + r) * DQK_ + c8);
        uint4 q1 = *(const uint4*)(qh + (size_t)(i0 + r + 16) * DQK_ + c8);
        uint4 q2 = *(const uint4*)(qh + (size_t)(i0 + r + 32) * DQK_ + c8);
        uint4 q3 = *(const uint4*)(qh + (size_t)(i0 + r + 48) * DQK_ + c8);
        *(uint4*)&Qs[r * 136 + c8]        = q0;
        *(uint4*)&Qs[(r + 16) * 136 + c8] = q1;
        *(uint4*)&Qs[(r + 32) * 136 + c8] = q2;
        *(uint4*)&Qs[(r + 48) * 136 + c8] = q3;
    }
    f32x8 accPV[4][2];
#pragma unroll
    for (int m = 0; m < 4; ++m) { accPV[m][0] = zero8(); accPV[m][1] = zero8(); }
    __syncthreads();

    const int mt = w >> 1, ntj = w & 1;  // this wave's S tile (of 4x2)
    for (int j0 = 0; j0 < i0 + 64; j0 += 32) {
        // batched global loads for K (32x128) and V (32x256) tiles
        {
            int r = tid >> 4, c8 = (tid & 15) * 8;
            uint4 k0r = *(const uint4*)(kh + (size_t)(j0 + r) * DQK_ + c8);
            uint4 k1r = *(const uint4*)(kh + (size_t)(j0 + r + 16) * DQK_ + c8);
            int rv = tid >> 5, cv = (tid & 31) * 8;
            U16x8 v0r, v1r, v2r, v3r;
            v0r.u4 = *(const uint4*)(vh + (size_t)(j0 + rv) * DV_ + cv);
            v1r.u4 = *(const uint4*)(vh + (size_t)(j0 + rv + 8) * DV_ + cv);
            v2r.u4 = *(const uint4*)(vh + (size_t)(j0 + rv + 16) * DV_ + cv);
            v3r.u4 = *(const uint4*)(vh + (size_t)(j0 + rv + 24) * DV_ + cv);
            if (j0 + 32 < i0 + 64) {
                __builtin_prefetch(kh + (size_t)(j0 + 32 + (tid >> 3)) * DQK_, 0, 0);
                __builtin_prefetch(vh + (size_t)(j0 + 32 + (tid >> 3)) * DV_, 0, 0);
            }
            *(uint4*)&Ks[r * 136 + c8]        = k0r;
            *(uint4*)&Ks[(r + 16) * 136 + c8] = k1r;
#pragma unroll
            for (int q = 0; q < 8; ++q) Vt[(cv + q) * GLS + rv]      = v0r.s[q];  // -> [n][j]
#pragma unroll
            for (int q = 0; q < 8; ++q) Vt[(cv + q) * GLS + rv + 8]  = v1r.s[q];
#pragma unroll
            for (int q = 0; q < 8; ++q) Vt[(cv + q) * GLS + rv + 16] = v2r.s[q];
#pragma unroll
            for (int q = 0; q < 8; ++q) Vt[(cv + q) * GLS + rv + 24] = v3r.s[q];
        }
        __syncthreads();

        // S = Q @ K^T  (this wave's 16x16 tile), K-loop over dqk=128
        f32x8 accS = zero8();
#pragma unroll
        for (int kc = 0; kc < 4; ++kc) {
            Frag a, bb;
            int qr = (mt * 16 + ln) * 136 + kc * 32;
            a.q[0] = *(uint4*)&Qs[qr + hiA];
            a.q[1] = *(uint4*)&Qs[qr + 16 + hiA];
            int krow = (ntj * 16 + ln) * 136 + kc * 32;
            bb.q[0] = *(uint4*)&Ks[krow + hiB];
            bb.q[1] = *(uint4*)&Ks[krow + hiB + 16];
            accS = wmma_bf16(a.v, bb.v, accS);
        }
        // decay + causal mask, pack to bf16 probs
        int jloc = ntj * 16 + ln;
#pragma unroll
        for (int v = 0; v < 8; ++v) {
            int i = i0 + mt * 16 + mb + v;
            int j = j0 + jloc;
            int e = i - j;
            float val = (e < 0) ? 0.0f : accS[v] * exp2f(l2g * (float)e);
            Ps[(mt * 16 + mb + v) * GLS + jloc] = f2bf(val);
        }
        __syncthreads();

        // PV: wave w owns v columns [32w, 32w+32)
        Frag bv[2];
#pragma unroll
        for (int nt2 = 0; nt2 < 2; ++nt2) {
            int n = w * 32 + nt2 * 16 + ln;
            bv[nt2].q[0] = *(uint4*)&Vt[n * GLS + hiB];
            bv[nt2].q[1] = *(uint4*)&Vt[n * GLS + hiB + 16];
        }
#pragma unroll
        for (int m = 0; m < 4; ++m) {
            Frag a;
            int pr = (m * 16 + ln) * GLS;
            a.q[0] = *(uint4*)&Ps[pr + hiA];
            a.q[1] = *(uint4*)&Ps[pr + 16 + hiA];
#pragma unroll
            for (int nt2 = 0; nt2 < 2; ++nt2) accPV[m][nt2] = wmma_bf16(a.v, bv[nt2].v, accPV[m][nt2]);
        }
        __syncthreads();
    }

#pragma unroll
    for (int m = 0; m < 4; ++m)
#pragma unroll
        for (int nt2 = 0; nt2 < 2; ++nt2)
#pragma unroll
            for (int v = 0; v < 8; ++v) {
                int i = i0 + m * 16 + mb + v;
                int col = h * DV_ + w * 32 + nt2 * 16 + ln;
                ret[((size_t)(b * T_ + i)) * VDIM_ + col] = accPV[m][nt2][v];
            }
}

// ---------------------------------------------------------------------------
// K5: curr_kv[b,h,kk,n] = sum_t k[t,kk]*v[t,n]*gamma^(T-1-t)  (WMMA GEMM 128x256xT)
// ---------------------------------------------------------------------------
__global__ __launch_bounds__(256, 1)
void kv_kernel(const unsigned short* __restrict__ kro, const unsigned short* __restrict__ vb,
               float* __restrict__ kvout) {
    __shared__ unsigned short Akv[128 * GLS];  // weighted k^T: [kk][t-chunk]
    __shared__ unsigned short Bkv[256 * GLS];  // v^T: [n][t-chunk]
    const int tid = threadIdx.x, lane = tid & 31, w = tid >> 5;
    const int ln = lane & 15;
    const int hiA = (lane >> 4) * 8, hiB = (lane >> 4) * 8, mb = (lane >> 4) * 8;
    const int bh = blockIdx.x, b = bh >> 3, h = bh & 7;
    const float l2g = head_log2gamma(h);
    const size_t hbase = (size_t)(b * NH_ + h) * T_;
    const unsigned short* kh = kro + hbase * DQK_;
    const unsigned short* vh = vb + hbase * DV_;

    f32x8 acc[8][2];
#pragma unroll
    for (int m = 0; m < 8; ++m) { acc[m][0] = zero8(); acc[m][1] = zero8(); }

    for (int j0 = 0; j0 < T_; j0 += 32) {
        {
            int jj = tid >> 4, c8 = (tid & 15) * 8;
            U16x8 k0r, k1r;
            k0r.u4 = *(const uint4*)(kh + (size_t)(j0 + jj) * DQK_ + c8);
            k1r.u4 = *(const uint4*)(kh + (size_t)(j0 + jj + 16) * DQK_ + c8);
            int rv = tid >> 5, cv = (tid & 31) * 8;
            U16x8 v0r, v1r, v2r, v3r;
            v0r.u4 = *(const uint4*)(vh + (size_t)(j0 + rv) * DV_ + cv);
            v1r.u4 = *(const uint4*)(vh + (size_t)(j0 + rv + 8) * DV_ + cv);
            v2r.u4 = *(const uint4*)(vh + (size_t)(j0 + rv + 16) * DV_ + cv);
            v3r.u4 = *(const uint4*)(vh + (size_t)(j0 + rv + 24) * DV_ + cv);
            if (j0 + 32 < T_) {
                __builtin_prefetch(kh + (size_t)(j0 + 32 + (tid >> 3)) * DQK_, 0, 0);
                __builtin_prefetch(vh + (size_t)(j0 + 32 + (tid >> 3)) * DV_, 0, 0);
            }
            float wt0 = exp2f(l2g * (float)(T_ - 1 - (j0 + jj)));
            float wt1 = exp2f(l2g * (float)(T_ - 1 - (j0 + jj + 16)));
#pragma unroll
            for (int q = 0; q < 8; ++q) Akv[(c8 + q) * GLS + jj]      = f2bf(bf2f(k0r.s[q]) * wt0);
#pragma unroll
            for (int q = 0; q < 8; ++q) Akv[(c8 + q) * GLS + jj + 16] = f2bf(bf2f(k1r.s[q]) * wt1);
#pragma unroll
            for (int q = 0; q < 8; ++q) Bkv[(cv + q) * GLS + rv]      = v0r.s[q];
#pragma unroll
            for (int q = 0; q < 8; ++q) Bkv[(cv + q) * GLS + rv + 8]  = v1r.s[q];
#pragma unroll
            for (int q = 0; q < 8; ++q) Bkv[(cv + q) * GLS + rv + 16] = v2r.s[q];
#pragma unroll
            for (int q = 0; q < 8; ++q) Bkv[(cv + q) * GLS + rv + 24] = v3r.s[q];
        }
        __syncthreads();

        Frag bv[2];
#pragma unroll
        for (int nt2 = 0; nt2 < 2; ++nt2) {
            int n = w * 32 + nt2 * 16 + ln;
            bv[nt2].q[0] = *(uint4*)&Bkv[n * GLS + hiB];
            bv[nt2].q[1] = *(uint4*)&Bkv[n * GLS + hiB + 16];
        }
#pragma unroll
        for (int m = 0; m < 8; ++m) {
            Frag a;
            int ar = (m * 16 + ln) * GLS;
            a.q[0] = *(uint4*)&Akv[ar + hiA];
            a.q[1] = *(uint4*)&Akv[ar + 16 + hiA];
#pragma unroll
            for (int nt2 = 0; nt2 < 2; ++nt2) acc[m][nt2] = wmma_bf16(a.v, bv[nt2].v, acc[m][nt2]);
        }
        __syncthreads();
    }

#pragma unroll
    for (int m = 0; m < 8; ++m)
#pragma unroll
        for (int nt2 = 0; nt2 < 2; ++nt2)
#pragma unroll
            for (int v = 0; v < 8; ++v) {
                int kk = m * 16 + mb + v;
                int n = w * 32 + nt2 * 16 + ln;
                kvout[((size_t)bh * DQK_ + kk) * DV_ + n] = acc[m][nt2][v];
            }
}

// ---------------------------------------------------------------------------
// K6: per-(b,t,head) group-norm over dv=256, in place
// ---------------------------------------------------------------------------
__global__ __launch_bounds__(256) void gn_kernel(float* __restrict__ ret) {
    __shared__ float red[256];
    const int tid = threadIdx.x;
    const int idx = blockIdx.x;
    const int h = idx & 7;
    const size_t bt = idx >> 3;
    float* p = ret + bt * VDIM_ + h * DV_;
    float v = p[tid];
    red[tid] = v;
    __syncthreads();
    for (int s = 128; s > 0; s >>= 1) {
        if (tid < s) red[tid] += red[tid + s];
        __syncthreads();
    }
    float mu = red[0] / 256.0f;
    __syncthreads();
    float d = v - mu;
    red[tid] = d * d;
    __syncthreads();
    for (int s = 128; s > 0; s >>= 1) {
        if (tid < s) red[tid] += red[tid + s];
        __syncthreads();
    }
    float var = red[0] / 256.0f;
    p[tid] = d * rsqrtf(var + 1e-5f);
}

// ---------------------------------------------------------------------------
// Host launcher
// ---------------------------------------------------------------------------
extern "C" void kernel_launch(void* const* d_in, const int* in_sizes, int n_in,
                              void* d_out, int out_size, void* d_ws, size_t ws_size,
                              hipStream_t stream) {
    (void)in_sizes; (void)n_in; (void)out_size; (void)ws_size;
    const float* x    = (const float*)d_in[0];
    const float* Wqkv = (const float*)d_in[1];
    const float* Wg   = (const float*)d_in[2];
    const float* Wp   = (const float*)d_in[3];
    const float* bp   = (const float*)d_in[4];

    char* ws = (char*)d_ws;
    const size_t OFF_XB   = 0;                       // bf16 x      [4096,1024]   8 MB
    const size_t OFF_WQKV = OFF_XB + 8388608;        // bf16 Wqkv   [1024,4096]   8 MB
    const size_t OFF_WG   = OFF_WQKV + 8388608;      // bf16 Wg     [1024,2048]   4 MB
    const size_t OFF_WP   = OFF_WG + 4194304;        // bf16 Wp     [2048,1024]   4 MB
    const size_t OFF_QKV  = OFF_WP + 4194304;        // f32 qkv     [4096,4096]  64 MB
    const size_t OFF_RET  = OFF_QKV;                 // f32 ret     [4096,2048]  (aliases qkv)
    const size_t OFF_H    = OFF_QKV + 33554432;      // bf16 gated  [4096,2048]  (aliases qkv)
    const size_t OFF_QRO  = OFF_QKV + 67108864;      // bf16 q rot  [2,8,2048,128]
    const size_t OFF_KRO  = OFF_QRO + 8388608;       // bf16 k rot
    const size_t OFF_VB   = OFF_KRO + 8388608;       // bf16 v      [2,8,2048,256]

    unsigned short* xb    = (unsigned short*)(ws + OFF_XB);
    unsigned short* wqkvb = (unsigned short*)(ws + OFF_WQKV);
    unsigned short* wgb   = (unsigned short*)(ws + OFF_WG);
    unsigned short* wpb   = (unsigned short*)(ws + OFF_WP);
    float*          qkv   = (float*)(ws + OFF_QKV);
    float*          ret   = (float*)(ws + OFF_RET);
    unsigned short* hbuf  = (unsigned short*)(ws + OFF_H);
    unsigned short* qro   = (unsigned short*)(ws + OFF_QRO);
    unsigned short* kro   = (unsigned short*)(ws + OFF_KRO);
    unsigned short* vbuf  = (unsigned short*)(ws + OFF_VB);

    float* out   = (float*)d_out;
    float* kvout = out + (size_t)B_ * T_ * HID_;  // second tuple output

    // 1) bf16 conversions
    cvt_kernel<<<4194304 / 256, 256, 0, stream>>>(x, xb, 4194304);
    cvt_kernel<<<4194304 / 256, 256, 0, stream>>>(Wqkv, wqkvb, 4194304);
    cvt_kernel<<<2097152 / 256, 256, 0, stream>>>(Wg, wgb, 2097152);
    cvt_kernel<<<2097152 / 256, 256, 0, stream>>>(Wp, wpb, 2097152);
    // 2) qkv = x @ Wqkv   [4096,1024]x[1024,4096]
    gemm_bf16_kernel<<<dim3(32, 32), 256, 0, stream>>>(xb, wqkvb, 4096, 4096, 1024, 0,
                                                       nullptr, qkv, nullptr);
    // 3) xPos rotate + per-head pack
    rotate_kernel<<<32768, 256, 0, stream>>>(qkv, qro, kro, vbuf);
    // 4) retention (causal decayed attention)
    retention_kernel<<<dim3(T_ / 64, B_ * NH_), 256, 0, stream>>>(qro, kro, vbuf, ret);
    // 5) curr_kv
    kv_kernel<<<B_ * NH_, 256, 0, stream>>>(kro, vbuf, kvout);
    // 6) group norm in place
    gn_kernel<<<B_ * T_ * NH_, 256, 0, stream>>>(ret);
    // 7) gated = silu(x @ Wg) * normed  -> bf16
    gemm_bf16_kernel<<<dim3(16, 32), 256, 0, stream>>>(xb, wgb, 4096, 2048, 1024, 1,
                                                       ret, nullptr, hbuf);
    // 8) out = gated @ Wp + bp
    gemm_bf16_kernel<<<dim3(8, 32), 256, 0, stream>>>(hbuf, wpb, 4096, 1024, 2048, 2,
                                                      bp, out, nullptr);
}
